// SpatialAttention_51324859187736
// MI455X (gfx1250) — matrix operand order repaired and checked by reading, not compiled, not run on hardware
//
#include <hip/hip_runtime.h>

#define B_   4
#define C_   512
#define CQ   64
#define N_   4096

typedef __attribute__((ext_vector_type(16))) _Float16 v16h;
typedef __attribute__((ext_vector_type(8)))  _Float16 v8h;
typedef __attribute__((ext_vector_type(8)))  float    v8f;

union AV { v16h v; v8h h[2]; };

// ---- CDNA5 async global->LDS path (probe via __has_builtin, sync fallback) ----
#if defined(__has_builtin)
#  if __has_builtin(__builtin_amdgcn_global_load_async_to_lds_b128)
#    define HAVE_ASYNC 1
#  endif
#endif
#ifndef HAVE_ASYNC
#  define HAVE_ASYNC 0
#endif

#if HAVE_ASYNC
typedef int v4i_t __attribute__((__vector_size__(4 * sizeof(int))));
typedef __attribute__((address_space(1))) v4i_t* gptr_v4i;
typedef __attribute__((address_space(3))) v4i_t* lptr_v4i;
#define ASYNC_CP16(srcp, dstp)                                              \
    __builtin_amdgcn_global_load_async_to_lds_b128(                         \
        (gptr_v4i)(srcp), (lptr_v4i)(dstp), 0, 0)
#if __has_builtin(__builtin_amdgcn_s_wait_asynccnt)
#define WAIT_ASYNC(n) __builtin_amdgcn_s_wait_asynccnt(n)
#else
#define WAIT_ASYNC(n) asm volatile("s_wait_asynccnt %0" ::"i"(n) : "memory")
#endif
#endif

static __device__ __forceinline__ v8f wmma_f16(v16h a, v16h b, v8f c) {
    return __builtin_amdgcn_wmma_f32_16x16x32_f16(false, a, false, b, (short)0, c,
                                                  false, false);
}

// A-matrix fragment (16x32 f16): caller passes pointer at row_base + k0 + (hi?8:0).
static __device__ __forceinline__ v16h ldA(const _Float16* p) {
    AV r;
    r.h[0] = *(const v8h*)p;
    r.h[1] = *(const v8h*)(p + 16);
    return r.v;
}

// B-matrix fragment (32x16 f16): caller passes pointer at col_base + k0 + (hi?8:0).
static __device__ __forceinline__ v16h ldB(const _Float16* p) {
    AV r;
    r.h[0] = *(const v8h*)p;
    r.h[1] = *(const v8h*)(p + 16);
    return r.v;
}

// ---------------- Kernel 0: weights f32 -> f16 ----------------
__global__ __launch_bounds__(256) void cvt_weights(
    const float* __restrict__ Wq, const float* __restrict__ Wk,
    const float* __restrict__ Wv,
    _Float16* __restrict__ Wq16, _Float16* __restrict__ Wk16,
    _Float16* __restrict__ Wv16)
{
    int id = blockIdx.x * 256 + threadIdx.x;
    if (id < CQ * C_) {
        Wq16[id] = (_Float16)Wq[id];
        Wk16[id] = (_Float16)Wk[id];
    }
    if (id < C_ * C_) Wv16[id] = (_Float16)Wv[id];
}

// ---------------- Kernel 1: projections q,k,v via WMMA ----------------
#define XPAD 520   // 512 + 8 halfwords LDS row stride (1040 B, 16B aligned)

__global__ __launch_bounds__(256) void proj_kernel(
    const float* __restrict__ x,
    const float* __restrict__ bq, const float* __restrict__ bk,
    const float* __restrict__ bv,
    const _Float16* __restrict__ Wq16, const _Float16* __restrict__ Wk16,
    const _Float16* __restrict__ Wv16,
    _Float16* __restrict__ qT, _Float16* __restrict__ kT,
    _Float16* __restrict__ vO)
{
    __shared__ __align__(16) _Float16 xT[64 * XPAD];

    const int tid = threadIdx.x;
    const int b  = blockIdx.x >> 6;
    const int i0 = (blockIdx.x & 63) << 6;

    for (int c = tid; c < C_; c += 256) {
        const float* xp = x + (((size_t)b * C_ + c) * N_ + i0);
#pragma unroll
        for (int i = 0; i < 64; i += 4) {
            float4 f = *(const float4*)(xp + i);
            xT[(i + 0) * XPAD + c] = (_Float16)f.x;
            xT[(i + 1) * XPAD + c] = (_Float16)f.y;
            xT[(i + 2) * XPAD + c] = (_Float16)f.z;
            xT[(i + 3) * XPAD + c] = (_Float16)f.w;
        }
    }
    __syncthreads();

    const int  w    = tid >> 5;
    const int  lane = tid & 31;
    const int  ln   = lane & 15;
    const bool hi   = lane >= 16;

    // ---- q (waves 0-3) / k (waves 4-7) ----
    {
        const int isub = w & 3;
        const _Float16* Wm = (w < 4) ? Wq16 : Wk16;
        const float*    bm = (w < 4) ? bq   : bk;
        _Float16*     outp = (w < 4) ? qT   : kT;

        v8f acc[4] = {};
        for (int ks = 0; ks < 16; ++ks) {
            const int k0 = ks * 32;
            v16h af = ldA(&xT[(isub * 16 + ln) * XPAD + k0 + (hi ? 8 : 0)]);
#pragma unroll
            for (int ct = 0; ct < 4; ++ct) {
                v16h bf = ldB(Wm + (size_t)(ct * 16 + ln) * C_ + k0 + (hi ? 8 : 0));
                acc[ct] = wmma_f16(af, bf, acc[ct]);
            }
        }
#pragma unroll
        for (int ct = 0; ct < 4; ++ct) {
            const int cq = ct * 16 + ln;
            const float bias = bm[cq];
#pragma unroll
            for (int r = 0; r < 8; ++r) {
                const int irow = i0 + isub * 16 + r + (hi ? 8 : 0);
                outp[((size_t)b * N_ + irow) * CQ + cq] = (_Float16)(acc[ct][r] + bias);
            }
        }
    }

    // ---- v: all 8 waves, wave w owns channels [w*64, w*64+64) ----
    {
        const int c0 = w * 64;
        v8f vacc[4][4] = {};
        for (int ks = 0; ks < 16; ++ks) {
            const int k0 = ks * 32;
            v16h af[4];
#pragma unroll
            for (int isub = 0; isub < 4; ++isub)
                af[isub] = ldA(&xT[(isub * 16 + ln) * XPAD + k0 + (hi ? 8 : 0)]);
#pragma unroll
            for (int ct = 0; ct < 4; ++ct) {
                v16h bf = ldB(Wv16 + (size_t)(c0 + ct * 16 + ln) * C_ + k0 + (hi ? 8 : 0));
#pragma unroll
                for (int isub = 0; isub < 4; ++isub)
                    vacc[isub][ct] = wmma_f16(af[isub], bf, vacc[isub][ct]);
            }
        }
#pragma unroll
        for (int ct = 0; ct < 4; ++ct) {
            const int c = c0 + ct * 16 + ln;
            const float bias = bv[c];
#pragma unroll
            for (int isub = 0; isub < 4; ++isub)
#pragma unroll
                for (int r = 0; r < 8; ++r) {
                    const int irow = i0 + isub * 16 + r + (hi ? 8 : 0);
                    vO[((size_t)b * C_ + c) * N_ + irow] = (_Float16)(vacc[isub][ct][r] + bias);
                }
        }
    }
}

// ---------------- Kernel 2: flash attention ----------------
#define JPAD 72    // 64 + 8 halfwords per LDS row (144 B, 16B aligned)

__global__ __launch_bounds__(256) void attn_kernel(
    const float* __restrict__ x, const float* __restrict__ gamma,
    const _Float16* __restrict__ qT, const _Float16* __restrict__ kT,
    const _Float16* __restrict__ vT, float* __restrict__ out)
{
    __shared__ __align__(16) _Float16 Kl[64 * JPAD];     //  9.2 KB, kT[j][c]
    __shared__ __align__(16) _Float16 Vl[C_ * JPAD];     // 73.7 KB, v[c][j]
    __shared__ __align__(16) _Float16 Pl[64 * JPAD];     //  9.2 KB, P[i][j]
    __shared__ float scal[64];

    const int  tid  = threadIdx.x;
    const int  w    = tid >> 5;
    const int  lane = tid & 31;
    const int  ln   = lane & 15;
    const bool hi   = lane >= 16;
    const int  b    = blockIdx.x >> 6;
    const int  i0   = (blockIdx.x & 63) << 6;

    v8f oacc[4][4] = {};

    const _Float16* qbase =
        qT + ((size_t)b * N_ + i0 + (w & 3) * 16 + ln) * CQ + (hi ? 8 : 0);
    v16h qa0 = ldA(qbase);
    v16h qa1 = ldA(qbase + 32);
    float m_run[8], l_run[8];
#pragma unroll
    for (int r = 0; r < 8; ++r) { m_run[r] = -1e30f; l_run[r] = 0.0f; }

    for (int j0 = 0; j0 < N_; j0 += 64) {
        __syncthreads();   // prior iteration done reading Kl/Vl/Pl

#if HAVE_ASYNC
        // Issue K chunks first (2 per wave), then V chunks (16 per wave).
        // Per-wave in-order completion => asynccnt<=16 means K tile landed.
        {
            int kid = w * 64 + lane;
#pragma unroll
            for (int k = 0; k < 2; ++k, kid += 32) {
                const int row = kid >> 3, cg = (kid & 7) << 3;   // 8 halfwords/chunk
                ASYNC_CP16(kT + ((size_t)b * N_ + j0 + row) * CQ + cg,
                           &Kl[row * JPAD + cg]);
            }
            int vid = w * 512 + lane;
#pragma unroll
            for (int k = 0; k < 16; ++k, vid += 32) {
                const int row = vid >> 3, cg = (vid & 7) << 3;
                ASYNC_CP16(vT + ((size_t)b * C_ + row) * N_ + j0 + cg,
                           &Vl[row * JPAD + cg]);
            }
        }
        WAIT_ASYNC(16);    // K tile complete; V tile still in flight
#else
        {
            const int row = tid >> 2, cg = (tid & 3) << 4;
            const _Float16* src = kT + ((size_t)b * N_ + j0 + row) * CQ + cg;
            v8h a0 = *(const v8h*)src;
            v8h a1 = *(const v8h*)(src + 8);
            *(v8h*)&Kl[row * JPAD + cg]     = a0;
            *(v8h*)&Kl[row * JPAD + cg + 8] = a1;
        }
#pragma unroll
        for (int k = 0; k < 8; ++k) {
            const int cid = tid + k * 256;
            const int row = cid >> 2, cg = (cid & 3) << 4;
            const _Float16* src = vT + ((size_t)b * C_ + row) * N_ + j0 + cg;
            v8h a0 = *(const v8h*)src;
            v8h a1 = *(const v8h*)(src + 8);
            *(v8h*)&Vl[row * JPAD + cg]     = a0;
            *(v8h*)&Vl[row * JPAD + cg + 8] = a1;
        }
#endif
        __syncthreads();   // K visible (async: V copy overlaps stage A below)

        // ---- stage A: scores + online softmax (waves 0-3) ----
        if (w < 4) {
            v8f s[4] = {};
#pragma unroll
            for (int jt = 0; jt < 4; ++jt) {
                const _Float16* kb = &Kl[(jt * 16 + ln) * JPAD + (hi ? 8 : 0)];
                s[jt] = wmma_f16(qa0, ldB(kb),      s[jt]);
                s[jt] = wmma_f16(qa1, ldB(kb + 32), s[jt]);
            }
#pragma unroll
            for (int r = 0; r < 8; ++r) {
                float mx = fmaxf(fmaxf(s[0][r], s[1][r]), fmaxf(s[2][r], s[3][r]));
#pragma unroll
                for (int off = 8; off >= 1; off >>= 1)
                    mx = fmaxf(mx, __shfl_xor(mx, off, 16));
                const float mn = fmaxf(m_run[r], mx);
                const float sc = __expf(m_run[r] - mn);
                m_run[r] = mn;
                float p0 = __expf(s[0][r] - mn);
                float p1 = __expf(s[1][r] - mn);
                float p2 = __expf(s[2][r] - mn);
                float p3 = __expf(s[3][r] - mn);
                float rs = (p0 + p1) + (p2 + p3);
#pragma unroll
                for (int off = 8; off >= 1; off >>= 1)
                    rs += __shfl_xor(rs, off, 16);
                l_run[r] = l_run[r] * sc + rs;

                const int row = w * 16 + r + (hi ? 8 : 0);
                if (ln == 0) scal[row] = sc;
                Pl[row * JPAD +  0 + ln] = (_Float16)p0;
                Pl[row * JPAD + 16 + ln] = (_Float16)p1;
                Pl[row * JPAD + 32 + ln] = (_Float16)p2;
                Pl[row * JPAD + 48 + ln] = (_Float16)p3;
            }
        }
#if HAVE_ASYNC
        WAIT_ASYNC(0);     // own V-tile asyncs complete before the barrier
#endif
        __syncthreads();   // P + full V tile visible

        // ---- O update: rescale then O += P x V^T (all 8 waves) ----
#pragma unroll
        for (int isub = 0; isub < 4; ++isub) {
            float scr[8];
#pragma unroll
            for (int r = 0; r < 8; ++r)
                scr[r] = scal[isub * 16 + r + (hi ? 8 : 0)];
#pragma unroll
            for (int ct = 0; ct < 4; ++ct)
#pragma unroll
                for (int r = 0; r < 8; ++r)
                    oacc[isub][ct][r] *= scr[r];
        }
#pragma unroll
        for (int kc = 0; kc < 2; ++kc) {
            v16h pa[4];
#pragma unroll
            for (int isub = 0; isub < 4; ++isub)
                pa[isub] = ldA(&Pl[(isub * 16 + ln) * JPAD + kc * 32 + (hi ? 8 : 0)]);
#pragma unroll
            for (int ct = 0; ct < 4; ++ct) {
                v16h bf = ldB(&Vl[(w * 64 + ct * 16 + ln) * JPAD + kc * 32 + (hi ? 8 : 0)]);
#pragma unroll
                for (int isub = 0; isub < 4; ++isub)
                    oacc[isub][ct] = wmma_f16(pa[isub], bf, oacc[isub][ct]);
            }
        }
    }

    // ---- epilogue: divide by softmax denom, gamma*O + x ----
    __syncthreads();
    if (w < 4) {
#pragma unroll
        for (int r = 0; r < 8; ++r) {
            const int row = w * 16 + r + (hi ? 8 : 0);
            if (ln == 0) scal[row] = 1.0f / l_run[r];
        }
    }
    __syncthreads();
    const float g = gamma[0];
#pragma unroll
    for (int isub = 0; isub < 4; ++isub) {
        float rr[8];
#pragma unroll
        for (int r = 0; r < 8; ++r)
            rr[r] = scal[isub * 16 + r + (hi ? 8 : 0)];
#pragma unroll
        for (int ct = 0; ct < 4; ++ct) {
            const int c = w * 64 + ct * 16 + ln;
#pragma unroll
            for (int r = 0; r < 8; ++r) {
                const int irow = i0 + isub * 16 + r + (hi ? 8 : 0);
                const size_t idx = ((size_t)b * C_ + c) * N_ + irow;
                out[idx] = g * oacc[isub][ct][r] * rr[r] + x[idx];
            }
        }
    }
}

extern "C" void kernel_launch(void* const* d_in, const int* in_sizes, int n_in,
                              void* d_out, int out_size, void* d_ws, size_t ws_size,
                              hipStream_t stream) {
    const float* x     = (const float*)d_in[0];
    const float* Wq    = (const float*)d_in[1];
    const float* bq    = (const float*)d_in[2];
    const float* Wk    = (const float*)d_in[3];
    const float* bk    = (const float*)d_in[4];
    const float* Wv    = (const float*)d_in[5];
    const float* bv    = (const float*)d_in[6];
    const float* gamma = (const float*)d_in[7];

    char* ws = (char*)d_ws;
    _Float16* qT   = (_Float16*)(ws);                                  // 2 MB
    _Float16* kT   = (_Float16*)(ws + (size_t)2  * 1024 * 1024);       // 2 MB
    _Float16* vT   = (_Float16*)(ws + (size_t)4  * 1024 * 1024);       // 16 MB
    _Float16* Wq16 = (_Float16*)(ws + (size_t)21 * 1024 * 1024);       // 64 KB
    _Float16* Wk16 = Wq16 + CQ * C_;                                   // 64 KB
    _Float16* Wv16 = Wk16 + CQ * C_;                                   // 512 KB

    cvt_weights<<<(C_ * C_) / 256, 256, 0, stream>>>(Wq, Wk, Wv, Wq16, Wk16, Wv16);
    proj_kernel<<<B_ * (N_ / 64), 256, 0, stream>>>(x, bq, bk, bv,
                                                    Wq16, Wk16, Wv16, qT, kT, vT);
    attn_kernel<<<B_ * (N_ / 64), 256, 0, stream>>>(x, gamma, qT, kT, vT,
                                                    (float*)d_out);
}
